// KANConvolution_44229573214883
// MI455X (gfx1250) — matrix operand order repaired and checked by reading, not compiled, not run on hardware
//
#include <hip/hip_runtime.h>

typedef _Float16 v16h __attribute__((ext_vector_type(16)));
typedef _Float16 v8h  __attribute__((ext_vector_type(8)));
typedef float    v8f  __attribute__((ext_vector_type(8)));
typedef unsigned int uint32x4 __attribute__((ext_vector_type(4)));
typedef int          int32x8  __attribute__((ext_vector_type(8)));
typedef int          int32x4  __attribute__((ext_vector_type(4)));

#define C_IN   32
#define C_OUT  64
#define BATCH  8
#define HW     32
#define L      (HW * HW)
#define KTAPS  9
#define MB     8
#define NPJ    9                    // values per (c,k): silu + 8 bases
#define RED    2592                 // KTAPS * 288
#define PWIDTH 288                  // C_IN * NPJ, per-pixel activation width
#define HP     34                   // halo-padded spatial dim
#define ROWS_PER_WG 64              // two oh rows
#define SPAN_BYTES  (32 * PWIDTH * 2)            // 18,432 B per oh row
#define TILE_BYTES  (ROWS_PER_WG * PWIDTH * 2)   // 36,864 B per tap tile
#define PITCH_B     (HP * PWIDTH * 2)            // 19,584 B between oh rows
#define ABUF_BYTES  (2 * TILE_BYTES)             // 73,728 B (double buffer)
#define OSM_BYTES   (16 * 16 * 17 * 4)           // 17,408 B
#define SMEM_BYTES  (ABUF_BYTES + OSM_BYTES)     // 91,136 B dynamic LDS

#if __has_builtin(__builtin_amdgcn_tensor_load_to_lds) && \
    __has_builtin(__builtin_amdgcn_s_wait_tensorcnt)
#define USE_TDM 1
#endif

__device__ __forceinline__ float silu_f(float t) {
    return t / (1.0f + __expf(-t));
}

// ---------------------------------------------------------------------------
// Kernel A: per x-element activations (one Cox-de Boor pass -> silu + 8 bases)
// Act layout: [b][ihp 0..33][iwp 0..33][c][j], j fastest. Halo = zeros.
// ---------------------------------------------------------------------------
__global__ void kan_precompute_act(const float* __restrict__ x,
                                   _Float16* __restrict__ act) {
    int idx = blockIdx.x * blockDim.x + threadIdx.x;
    const int total = BATCH * HP * HP * C_IN;
    if (idx >= total) return;
    int c   = idx & 31;
    int pix = idx >> 5;
    int iwp = pix % HP;
    int t2  = pix / HP;
    int ihp = t2 % HP;
    int b   = t2 / HP;
    _Float16* dst = act + ((size_t)pix * C_IN + c) * NPJ;

    if (ihp == 0 || ihp == HP - 1 || iwp == 0 || iwp == HP - 1) {
#pragma unroll
        for (int j = 0; j < NPJ; ++j) dst[j] = (_Float16)0.0f;
        return;
    }
    float t = x[(((b * C_IN) + c) * HW + (ihp - 1)) * HW + (iwp - 1)];

    const float h = 0.4f;                  // grid: g(i) = -1 + (i-3)*h
    float Bv[11];
#pragma unroll
    for (int i = 0; i < 11; ++i) {
        float g0 = -1.0f + (float)(i - 3) * h;
        float g1 = -1.0f + (float)(i - 2) * h;
        Bv[i] = (t >= g0 && t < g1) ? 1.0f : 0.0f;
    }
#pragma unroll
    for (int p = 1; p <= 3; ++p) {
        float inv = 1.0f / ((float)p * h);
#pragma unroll
        for (int i = 0; i < 11 - p; ++i) {
            float gi  = -1.0f + (float)(i - 3) * h;
            float gip = -1.0f + (float)(i + p - 2) * h;   // g(i+p+1)
            Bv[i] = ((t - gi) * Bv[i] + (gip - t) * Bv[i + 1]) * inv;
        }
    }
    dst[0] = (_Float16)silu_f(t);
#pragma unroll
    for (int m = 0; m < MB; ++m) dst[1 + m] = (_Float16)Bv[m];
}

// ---------------------------------------------------------------------------
// Kernel B: pack weights f16; r = k*288 + c*9 + j  (matches Act ordering)
// ---------------------------------------------------------------------------
__global__ void kan_pack_weights(const float* __restrict__ coef,
                                 const float* __restrict__ scale_base,
                                 const float* __restrict__ scale_sp,
                                 _Float16* __restrict__ wpk) {
    int idx = blockIdx.x * blockDim.x + threadIdx.x;
    if (idx >= C_OUT * RED) return;
    int o   = idx / RED;
    int r   = idx - o * RED;
    int k   = r / PWIDTH;
    int rem = r - k * PWIDTH;
    int c   = rem / NPJ;
    int j   = rem - c * NPJ;
    int ock = (o * C_IN + c) * KTAPS + k;
    float w = (j == 0) ? scale_base[ock]
                       : scale_sp[ock] * coef[ock * MB + (j - 1)];
    wpk[idx] = (_Float16)w;
}

#ifdef USE_TDM
// 2D TDM: 2 spans of 18,432 B (one per oh row), pitch 19,584 B, into LDS.
__device__ __forceinline__ void tdm_issue(const _Float16* src, unsigned ldsoff) {
    unsigned long long ga = (unsigned long long)(uintptr_t)src;
    const unsigned n8  = SPAN_BYTES / 8;     // 2304: tile_dim0 (8B units)
    const unsigned st8 = PITCH_B / 8;        // 2448: tensor_dim0_stride
    uint32x4 g0;
    g0[0] = 1u;                                          // count=1
    g0[1] = ldsoff;                                      // lds_addr
    g0[2] = (unsigned)(ga & 0xffffffffu);                // global_addr lo
    g0[3] = (unsigned)((ga >> 32) & 0x01ffffffu) | (2u << 30);  // type=2
    int32x8 g1;
    g1[0] = (int)(3u << 16);                 // data_size = 8B, mask=0
    g1[1] = (int)((st8 & 0xffffu) << 16);    // tensor_dim0 lo16 (=2448)
    g1[2] = (int)((st8 >> 16) | (2u << 16)); // tensor_dim0 hi | tensor_dim1=2
    g1[3] = (int)(n8 << 16);                 // tile_dim0 = 2304
    g1[4] = 2;                               // tile_dim1 = 2, tile_dim2 = 0
    g1[5] = (int)st8;                        // tensor_dim0_stride lo32
    g1[6] = 0;
    g1[7] = 0;
    int32x4 gz = {0, 0, 0, 0};
#if defined(__clang_major__) && __clang_major__ >= 23
    int32x8 gz8 = {0, 0, 0, 0, 0, 0, 0, 0};
    __builtin_amdgcn_tensor_load_to_lds(g0, g1, gz, gz, gz8, 0);
#else
    __builtin_amdgcn_tensor_load_to_lds(g0, g1, gz, gz, 0);
#endif
}
#endif

// ---------------------------------------------------------------------------
// Kernel C: WMMA GEMM. WG = 8 waves; 64 rows (two oh rows) x 64 out channels.
// Wave w: col tile (w&3), rows (w>>2)*32 + {0,16}. Per K-chunk: one B
// fragment, two A fragments, two WMMAs. TDM double-buffered across taps.
// ---------------------------------------------------------------------------
__launch_bounds__(256)
__global__ void kan_conv_wmma(const _Float16* __restrict__ act,
                              const _Float16* __restrict__ wpk,
                              const float* __restrict__ bias,
                              float* __restrict__ out) {
    extern __shared__ char smem[];
    _Float16* Asm = (_Float16*)smem;                     // [2][64][288]
    float*    Osm = (float*)(smem + ABUF_BYTES);         // [16][16][17]

    const int tid     = threadIdx.x;
    const int wave    = tid >> 5;
    const int lane    = tid & 31;
    const int hh      = lane >> 4;
    const int ln16    = lane & 15;
    const int rowPair = wave >> 2;   // 0..1 -> rows 0..31 / 32..63
    const int colTile = wave & 3;
    const int wgRow   = blockIdx.x * ROWS_PER_WG;
    const int bIdx    = wgRow >> 10;
    const int oh0     = (wgRow & 1023) >> 5;

    v8f acc0 = {}, acc1 = {};

#ifdef USE_TDM
    if (wave == 0) {   // prologue: tap 0 -> buffer 0
        const _Float16* s0 = act + ((size_t)((bIdx * HP + oh0) * HP + 0)) * PWIDTH;
        tdm_issue(s0, (unsigned)(uintptr_t)&Asm[0]);
    }
#endif

    for (int k = 0; k < KTAPS; ++k) {
        const int buf = k & 1;
        __syncthreads();   // prior consumption done; buf[(k+1)&1] is free
#ifdef USE_TDM
        if (wave == 0) {
            if (k < KTAPS - 1) {
                const int kn = k + 1;
                const _Float16* sn = act +
                    ((size_t)((bIdx * HP + oh0 + kn / 3) * HP + kn % 3)) * PWIDTH;
                tdm_issue(sn, (unsigned)(uintptr_t)&Asm[(size_t)(kn & 1) *
                                                        (TILE_BYTES / 2)]);
                __builtin_amdgcn_s_wait_tensorcnt(1);   // tap k has landed
            } else {
                __builtin_amdgcn_s_wait_tensorcnt(0);
            }
        }
#else
        {
            const _Float16* src = act +
                ((size_t)((bIdx * HP + oh0 + k / 3) * HP + k % 3)) * PWIDTH;
            const v8h* s = reinterpret_cast<const v8h*>(src);
            v8h* d = reinterpret_cast<v8h*>(&Asm[(size_t)buf * (TILE_BYTES / 2)]);
            const int perSpan = SPAN_BYTES / 16;         // 1152 v8h per oh row
            const int pitchV  = PITCH_B / 16;            // 1224 v8h pitch
            for (int i = tid; i < TILE_BYTES / 16; i += 256)
                d[i] = s[(i / perSpan) * pitchV + (i % perSpan)];
        }
#endif
        __syncthreads();   // tile k visible to all waves

        const _Float16* abase = &Asm[(size_t)buf * (TILE_BYTES / 2)];
#pragma unroll
        for (int q = 0; q < 9; ++q) {
            // B fragment (shared by both row tiles)
            const _Float16* wp = wpk + (size_t)(colTile * 16 + ln16) * RED
                                 + k * PWIDTH + q * 32 + hh * 8;
            const v8h blo = *reinterpret_cast<const v8h*>(wp);
            const v8h bhi = *reinterpret_cast<const v8h*>(wp + 16);
            v16h bm;
#pragma unroll
            for (int i = 0; i < 8; ++i) { bm[i] = blo[i]; bm[i + 8] = bhi[i]; }

            // A fragment, row tile 0 of this wave
            const int r0 = rowPair * 32 + ln16;
            const v8h a0lo = *reinterpret_cast<const v8h*>(
                abase + r0 * PWIDTH + q * 32 + hh * 8);
            const v8h a0hi = *reinterpret_cast<const v8h*>(
                abase + r0 * PWIDTH + q * 32 + 16 + hh * 8);
            v16h a0;
#pragma unroll
            for (int i = 0; i < 8; ++i) { a0[i] = a0lo[i]; a0[i + 8] = a0hi[i]; }
            acc0 = __builtin_amdgcn_wmma_f32_16x16x32_f16(
                false, a0, false, bm, (short)0, acc0, false, false);

            // A fragment, row tile 1
            const int r1 = r0 + 16;
            const v8h a1lo = *reinterpret_cast<const v8h*>(
                abase + r1 * PWIDTH + q * 32 + hh * 8);
            const v8h a1hi = *reinterpret_cast<const v8h*>(
                abase + r1 * PWIDTH + q * 32 + 16 + hh * 8);
            v16h a1;
#pragma unroll
            for (int i = 0; i < 8; ++i) { a1[i] = a1lo[i]; a1[i + 8] = a1hi[i]; }
            acc1 = __builtin_amdgcn_wmma_f32_16x16x32_f16(
                false, a1, false, bm, (short)0, acc1, false, false);
        }
    }

    // ---- transpose accumulators through LDS for coalesced stores ----
#pragma unroll
    for (int i = 0; i < 8; ++i) {
        Osm[(((wave * 2 + 0) * 16) + hh * 8 + i) * 17 + ln16] = acc0[i];
        Osm[(((wave * 2 + 1) * 16) + hh * 8 + i) * 17 + ln16] = acc1[i];
    }
    __syncthreads();

#pragma unroll
    for (int e = 0; e < 16; ++e) {
        int idx = e * 256 + tid;
        int tl  = idx >> 8;              // 0..15: wave*2 + half
        int oo  = (idx >> 4) & 15;
        int mm  = idx & 15;
        int wv  = tl >> 1;
        int hf  = tl & 1;
        int o   = (wv & 3) * 16 + oo;
        int grow = wgRow + (wv >> 2) * 32 + hf * 16 + mm;
        int b   = grow >> 10;
        int l   = grow & 1023;
        out[((size_t)(b * C_OUT + o)) * L + l] =
            Osm[(tl * 16 + mm) * 17 + oo] + bias[o];
    }
}

// ---------------------------------------------------------------------------
extern "C" void kernel_launch(void* const* d_in, const int* in_sizes, int n_in,
                              void* d_out, int out_size, void* d_ws, size_t ws_size,
                              hipStream_t stream) {
    (void)in_sizes; (void)n_in; (void)out_size; (void)ws_size;
    const float* x          = (const float*)d_in[0];
    const float* coef       = (const float*)d_in[1];
    const float* scale_base = (const float*)d_in[2];
    const float* scale_sp   = (const float*)d_in[3];
    const float* bias       = (const float*)d_in[4];

    _Float16* wpk = (_Float16*)d_ws;                        // 331,776 B
    _Float16* act = (_Float16*)((char*)d_ws + 0x52000);     // 5,326,848 B

    const int nAct = BATCH * HP * HP * C_IN;                // 295,936 sites
    kan_precompute_act<<<(nAct + 255) / 256, 256, 0, stream>>>(x, act);

    const int nW = C_OUT * RED;
    kan_pack_weights<<<(nW + 255) / 256, 256, 0, stream>>>(
        coef, scale_base, scale_sp, wpk);

    const int nBlocks = (BATCH * L) / ROWS_PER_WG;          // 128 workgroups
    kan_conv_wmma<<<nBlocks, 256, SMEM_BYTES, stream>>>(
        act, wpk, bias, (float*)d_out);
}
